// SelfAttention_27187142984227
// MI455X (gfx1250) — matrix operand order repaired and checked by reading, not compiled, not run on hardware
//
#include <hip/hip_runtime.h>
#include <hip/hip_bf16.h>
#include <stdint.h>

// ---------------------------------------------------------------------------
// Self-attention (B=1, S=4096, H=1024, NH=16, HD=64) for MI455X / gfx1250.
// bf16 WMMA everywhere; flash-softmax attention (no 2x1GB score tensors);
// causal mask applied analytically (64MB mask input never read).
// Round 3: structural ping-pong (2x unrolled) GEMM k-loops -- no buffer
// rotation movs, no WMMA hazard v_nops; global_prefetch of next K/V chunk
// in the attention loop.
// ---------------------------------------------------------------------------

#define S_LEN 4096
#define H_DIM 1024
#define NHEAD 16
#define HD    64

typedef __attribute__((ext_vector_type(16))) __bf16 v16bf;
typedef __attribute__((ext_vector_type(8)))  float  v8f;

union FragAB {
    uint4          q[2];
    v16bf          v;
    unsigned short s[16];
};

__device__ __forceinline__ unsigned short f2bf(float f) {
    union { float f; unsigned int u; } c; c.f = f;
    unsigned int u = c.u;
    return (unsigned short)((u + 0x7FFFu + ((u >> 16) & 1u)) >> 16); // RNE
}

// A fragment: 16x32 bf16 (MxK), row-major source with leading dim ld.
// lanes 0-15: M=lane, elems[0..7]=K0..7,  elems[8..15]=K16..23
// lanes16-31: M=lane-16, elems[0..7]=K8..15, elems[8..15]=K24..31
__device__ __forceinline__ v16bf load_a_frag(const unsigned short* A, int ld,
                                             int m0, int k0) {
    int lane = threadIdx.x & 31;
    int m    = lane & 15;
    int kb   = (lane < 16) ? 0 : 8;
    const unsigned short* p = A + (size_t)(m0 + m) * ld + k0 + kb;
    FragAB f;
    f.q[0] = *reinterpret_cast<const uint4*>(p);
    f.q[1] = *reinterpret_cast<const uint4*>(p + 16);
    return f.v;
}

// B fragment: 32x16 bf16 (KxN). Source is N-major ("Bt"): Bt[n][k], ld = K dim.
// lanes 0-15: N=lane, elems[0..7]=K0..7, elems[8..15]=K16..23; lanes 16-31: N=lane-16, K8..15, K24..31.
__device__ __forceinline__ v16bf load_b_frag(const unsigned short* Bt, int ld,
                                             int n0, int k0) {
    int lane = threadIdx.x & 31;
    int n    = lane & 15;
    int kb   = (lane < 16) ? 0 : 8;
    const unsigned short* p = Bt + (size_t)(n0 + n) * ld + k0 + kb;
    FragAB f;
    f.q[0] = *reinterpret_cast<const uint4*>(p);
    f.q[1] = *reinterpret_cast<const uint4*>(p + 16);
    return f.v;
}

__device__ __forceinline__ v8f wmma_bf16(v16bf a, v16bf b, v8f c) {
    return __builtin_amdgcn_wmma_f32_16x16x32_bf16(
        /*neg_a=*/false, a, /*neg_b=*/false, b,
        /*c_mod=*/(short)0, c, /*reuse_a=*/false, /*reuse_b=*/false);
}

// ---------------------------------------------------------------------------
// Conversion kernels
// ---------------------------------------------------------------------------
__global__ void cvt_bf16_kernel(const float* __restrict__ x,
                                unsigned short* __restrict__ y, int n) {
    int i = blockIdx.x * blockDim.x + threadIdx.x;
    int stride = gridDim.x * blockDim.x;
    for (; i < n; i += stride) y[i] = f2bf(x[i]);
}

// x: [K][N] row-major f32  ->  y: [N][K] bf16 (transposed)
__global__ void cvt_transpose_kernel(const float* __restrict__ x,
                                     unsigned short* __restrict__ y,
                                     int K, int N) {
    int i = blockIdx.x * blockDim.x + threadIdx.x;
    int total = K * N;
    int stride = gridDim.x * blockDim.x;
    for (; i < total; i += stride) {
        int n = i / K, k = i % K;
        y[i] = f2bf(x[(size_t)k * N + n]);
    }
}

// ---------------------------------------------------------------------------
// Shared GEMM inner machinery: 32(M) x 64(N) per wave, ping-pong pipelined
// k-loop over K=H_DIM, 8 WMMAs per 32-wide k-step, no buffer-rotation movs.
// ---------------------------------------------------------------------------
#define GEMM_LOAD(AF, BF, Abase, Bbase, kk)                                   \
    do {                                                                      \
        AF[0] = load_a_frag(Abase, H_DIM, m0,      (kk));                     \
        AF[1] = load_a_frag(Abase, H_DIM, m0 + 16, (kk));                     \
        BF[0] = load_b_frag(Bbase, H_DIM, n0,      (kk));                     \
        BF[1] = load_b_frag(Bbase, H_DIM, n0 + 16, (kk));                     \
        BF[2] = load_b_frag(Bbase, H_DIM, n0 + 32, (kk));                     \
        BF[3] = load_b_frag(Bbase, H_DIM, n0 + 48, (kk));                     \
    } while (0)

#define GEMM_MMA(AF, BF)                                                      \
    do {                                                                      \
        _Pragma("unroll")                                                     \
        for (int mi = 0; mi < 2; ++mi) {                                      \
            _Pragma("unroll")                                                 \
            for (int t = 0; t < 4; ++t)                                       \
                acc[mi][t] = wmma_bf16(AF[mi], BF[t], acc[mi][t]);            \
        }                                                                     \
    } while (0)

#define GEMM_BODY(Abase, Bbase)                                               \
    v8f acc[2][4] = {};                                                       \
    v16bf a0[2], b0[4], a1[2], b1[4];                                         \
    GEMM_LOAD(a0, b0, Abase, Bbase, 0);                                       \
    int k = 0;                                                                \
    for (; k + 64 < H_DIM; k += 64) {                                         \
        GEMM_LOAD(a1, b1, Abase, Bbase, k + 32);  /* issue, then compute */   \
        GEMM_MMA(a0, b0);                         /* covers load latency */   \
        GEMM_LOAD(a0, b0, Abase, Bbase, k + 64);                              \
        GEMM_MMA(a1, b1);                                                     \
    }                                                                         \
    GEMM_LOAD(a1, b1, Abase, Bbase, k + 32);      /* k == H_DIM - 64 */       \
    GEMM_MMA(a0, b0);                                                         \
    GEMM_MMA(a1, b1);

// ---------------------------------------------------------------------------
// QKV projection: [S,H] x [H,3H] + bias -> Q (scaled by 1/sqrt(HD)), K, Vt.
// ---------------------------------------------------------------------------
__global__ void __launch_bounds__(256)
qkv_gemm_kernel(const unsigned short* __restrict__ Xb,     // [S][H] bf16
                const unsigned short* __restrict__ WqkvT,  // [3H][H] bf16
                const float* __restrict__ b_qkv,           // [3H]
                unsigned short* __restrict__ Qb,           // [NH][S][HD]
                unsigned short* __restrict__ Kb,           // [NH][S][HD]
                unsigned short* __restrict__ Vt) {         // [NH][HD][S]
    const int NSTRIP = (3 * H_DIM) / 64;                   // 48
    int wave = (blockIdx.x * blockDim.x + threadIdx.x) >> 5;
    int mt = wave / NSTRIP, nsx = wave % NSTRIP;
    int m0 = mt * 32, n0 = nsx * 64;

    GEMM_BODY(Xb, WqkvT);

    int lane = threadIdx.x & 31;
    #pragma unroll
    for (int mi = 0; mi < 2; ++mi) {
        int rbase = m0 + mi * 16 + ((lane < 16) ? 0 : 8);
        #pragma unroll
        for (int t = 0; t < 4; ++t) {
            int n       = n0 + t * 16 + (lane & 15);
            float bias  = b_qkv[n];
            int section = n / H_DIM;       // 0=Q 1=K 2=V
            int hcol    = n % H_DIM;
            int head    = hcol / HD, d = hcol % HD;
            float scale = (section == 0) ? 0.125f : 1.0f;  // 1/sqrt(64)
            #pragma unroll
            for (int r = 0; r < 8; ++r) {
                unsigned short h = f2bf((acc[mi][t][r] + bias) * scale);
                int row = rbase + r;
                if (section == 0)
                    Qb[((size_t)(head * S_LEN + row)) * HD + d] = h;
                else if (section == 1)
                    Kb[((size_t)(head * S_LEN + row)) * HD + d] = h;
                else
                    Vt[((size_t)(head * HD + d)) * S_LEN + row] = h;
            }
        }
    }
}

// ---------------------------------------------------------------------------
// Flash attention: one wave per (head, 32-query strip). Causal.
// K/V fragments loaded once per 32-key chunk feed 16 WMMAs.
// ---------------------------------------------------------------------------
__global__ void __launch_bounds__(256)
attn_kernel(const unsigned short* __restrict__ Qb,   // [NH][S][HD]
            const unsigned short* __restrict__ Kb,   // [NH][S][HD]
            const unsigned short* __restrict__ Vt,   // [NH][HD][S]
            unsigned short* __restrict__ Ctx) {      // [S][H] bf16
    __shared__ __align__(16) unsigned short lds_p[8][2][16 * 32];
    const int QSTRIPS = S_LEN / 32;                  // 128
    int wib   = threadIdx.x >> 5;
    int gwave = blockIdx.x * 8 + wib;
    int head  = gwave / QSTRIPS;
    int q0    = (gwave % QSTRIPS) * 32;
    int lane  = threadIdx.x & 31;
    int rbase = (lane < 16) ? 0 : 8;

    const unsigned short* Qh = Qb + (size_t)head * S_LEN * HD;
    const unsigned short* Kh = Kb + (size_t)head * S_LEN * HD;
    const unsigned short* Vh = Vt + (size_t)head * HD * S_LEN;

    // loop-invariant Q fragments: 2 row tiles x 2 k-chunks
    v16bf aQ[2][2];
    #pragma unroll
    for (int t = 0; t < 2; ++t) {
        aQ[t][0] = load_a_frag(Qh, HD, q0 + t * 16, 0);
        aQ[t][1] = load_a_frag(Qh, HD, q0 + t * 16, 32);
    }

    v8f O[2][4] = {};
    float mrun[2][8], lrun[2][8];
    #pragma unroll
    for (int t = 0; t < 2; ++t)
        #pragma unroll
        for (int i = 0; i < 8; ++i) { mrun[t][i] = -3.0e30f; lrun[t][i] = 0.0f; }

    int q_end = q0 + 32;                        // causal: keys < q_end
    for (int kb = 0; kb < q_end; kb += 32) {
        // hoist ALL fragment loads: V loads overlap scores WMMAs + softmax
        v16bf bK[2][2], bV[4];
        bK[0][0] = load_b_frag(Kh, HD, kb,      0);
        bK[0][1] = load_b_frag(Kh, HD, kb,      32);
        bK[1][0] = load_b_frag(Kh, HD, kb + 16, 0);
        bK[1][1] = load_b_frag(Kh, HD, kb + 16, 32);
        #pragma unroll
        for (int c = 0; c < 4; ++c) bV[c] = load_b_frag(Vh, S_LEN, c * 16, kb);

        #pragma unroll
        for (int t = 0; t < 2; ++t) {
            // scores: S[16x32] = Q[16x64] . K^T, two 16x16 subtiles
            v8f s0 = {}, s1 = {};
            s0 = wmma_bf16(aQ[t][0], bK[0][0], s0);
            s0 = wmma_bf16(aQ[t][1], bK[0][1], s0);
            s1 = wmma_bf16(aQ[t][0], bK[1][0], s1);
            s1 = wmma_bf16(aQ[t][1], bK[1][1], s1);

            // causal mask + online softmax (rows=VGPR idx, cols=lanes)
            #pragma unroll
            for (int i = 0; i < 8; ++i) {
                int row   = q0 + t * 16 + rbase + i;
                int key_a = kb + (lane & 15);
                if (key_a > row)      s0[i] = -10000.0f;
                if (key_a + 16 > row) s1[i] = -10000.0f;

                float mx = fmaxf(s0[i], s1[i]);
                #pragma unroll
                for (int off = 1; off < 16; off <<= 1)
                    mx = fmaxf(mx, __shfl_xor(mx, off, 32));
                float mnew  = fmaxf(mrun[t][i], mx);
                float alpha = __expf(mrun[t][i] - mnew);
                mrun[t][i] = mnew;
                float p0 = __expf(s0[i] - mnew);
                float p1 = __expf(s1[i] - mnew);
                s0[i] = p0; s1[i] = p1;
                float rs = p0 + p1;
                #pragma unroll
                for (int off = 1; off < 16; off <<= 1)
                    rs += __shfl_xor(rs, off, 32);
                lrun[t][i] = lrun[t][i] * alpha + rs;
                O[t][0][i] *= alpha; O[t][1][i] *= alpha;
                O[t][2][i] *= alpha; O[t][3][i] *= alpha;
            }

            // stage P (C-layout -> A-layout) through wave-private LDS
            unsigned short* pw = lds_p[wib][t];
            #pragma unroll
            for (int i = 0; i < 8; ++i) {
                int row = rbase + i, col = lane & 15;
                pw[row * 32 + col]      = f2bf(s0[i]);
                pw[row * 32 + col + 16] = f2bf(s1[i]);
            }
        }

        // prefetch next key chunk (global_prefetch_b8; no counters, no VGPRs)
        if (kb + 32 < q_end) {
            int nlane = lane & 15;
            __builtin_prefetch(Kh + (size_t)(kb + 32 + nlane) * HD, 0, 3);
            __builtin_prefetch(Kh + (size_t)(kb + 48 + nlane) * HD, 0, 3);
            __builtin_prefetch(Vh + (size_t)nlane * S_LEN + kb + 32, 0, 3);
            __builtin_prefetch(Vh + (size_t)(48 + nlane) * S_LEN + kb + 32, 0, 3);
        }

        // O[t][16x64] += P[t][16x32] . V[32x64] (V stored transposed)
        #pragma unroll
        for (int t = 0; t < 2; ++t) {
            int m  = lane & 15;
            int kB = (lane < 16) ? 0 : 8;
            const unsigned short* pp = lds_p[wib][t] + m * 32 + kB;
            FragAB f;
            f.q[0] = *reinterpret_cast<const uint4*>(pp);
            f.q[1] = *reinterpret_cast<const uint4*>(pp + 16);
            v16bf aP = f.v;
            #pragma unroll
            for (int c = 0; c < 4; ++c)
                O[t][c] = wmma_bf16(aP, bV[c], O[t][c]);
        }
    }

    // normalize and store ctx as bf16 [S][H]
    int colbase = head * HD + (lane & 15);
    #pragma unroll
    for (int t = 0; t < 2; ++t)
        #pragma unroll
        for (int i = 0; i < 8; ++i) {
            float inv = 1.0f / lrun[t][i];
            size_t base = (size_t)(q0 + t * 16 + rbase + i) * H_DIM + colbase;
            Ctx[base]      = f2bf(O[t][0][i] * inv);
            Ctx[base + 16] = f2bf(O[t][1][i] * inv);
            Ctx[base + 32] = f2bf(O[t][2][i] * inv);
            Ctx[base + 48] = f2bf(O[t][3][i] * inv);
        }
}

// ---------------------------------------------------------------------------
// Output projection: [S,H] x [H,H] + bias -> f32 out.
// ---------------------------------------------------------------------------
__global__ void __launch_bounds__(256)
out_gemm_kernel(const unsigned short* __restrict__ Ctx,    // [S][H] bf16
                const unsigned short* __restrict__ WoutT,  // [H][H] bf16 (T)
                const float* __restrict__ b_out,
                float* __restrict__ Out) {                 // [S][H] f32
    const int NSTRIP = H_DIM / 64;                         // 16
    int wave = (blockIdx.x * blockDim.x + threadIdx.x) >> 5;
    int mt = wave / NSTRIP, nsx = wave % NSTRIP;
    int m0 = mt * 32, n0 = nsx * 64;

    GEMM_BODY(Ctx, WoutT);

    int lane = threadIdx.x & 31;
    #pragma unroll
    for (int mi = 0; mi < 2; ++mi) {
        int rbase = m0 + mi * 16 + ((lane < 16) ? 0 : 8);
        #pragma unroll
        for (int t = 0; t < 4; ++t) {
            int n = n0 + t * 16 + (lane & 15);
            float bias = b_out[n];
            #pragma unroll
            for (int r = 0; r < 8; ++r)
                Out[(size_t)(rbase + r) * H_DIM + n] = acc[mi][t][r] + bias;
        }
    }
}

// ---------------------------------------------------------------------------
extern "C" void kernel_launch(void* const* d_in, const int* in_sizes, int n_in,
                              void* d_out, int out_size, void* d_ws,
                              size_t ws_size, hipStream_t stream) {
    (void)in_sizes; (void)n_in; (void)out_size; (void)ws_size;
    const float* hidden = (const float*)d_in[0];
    // d_in[1] = causal mask: never read (tril applied analytically)
    const float* W_qkv  = (const float*)d_in[2];
    const float* b_qkv  = (const float*)d_in[3];
    const float* W_out  = (const float*)d_in[4];
    const float* b_out  = (const float*)d_in[5];
    float*       out    = (float*)d_out;

    char* ws = (char*)d_ws;
    size_t off = 0;
    unsigned short* Xb    = (unsigned short*)(ws + off); off += (size_t)S_LEN * H_DIM * 2;
    unsigned short* WqkvT = (unsigned short*)(ws + off); off += (size_t)3 * H_DIM * H_DIM * 2;
    unsigned short* WoutT = (unsigned short*)(ws + off); off += (size_t)H_DIM * H_DIM * 2;
    unsigned short* Qb    = (unsigned short*)(ws + off); off += (size_t)NHEAD * S_LEN * HD * 2;
    unsigned short* Kb    = (unsigned short*)(ws + off); off += (size_t)NHEAD * S_LEN * HD * 2;
    unsigned short* Vt    = (unsigned short*)(ws + off); off += (size_t)NHEAD * HD * S_LEN * 2;
    unsigned short* Ctx   = (unsigned short*)(ws + off); off += (size_t)S_LEN * H_DIM * 2;
    // total workspace: 48 MiB (fits MI455X's 192 MB L2)

    cvt_bf16_kernel<<<1024, 256, 0, stream>>>(hidden, Xb, S_LEN * H_DIM);
    cvt_transpose_kernel<<<1024, 256, 0, stream>>>(W_qkv, WqkvT, H_DIM, 3 * H_DIM);
    cvt_transpose_kernel<<<1024, 256, 0, stream>>>(W_out, WoutT, H_DIM, H_DIM);

    // 128 M-strips x 48 N-strips = 6144 waves, 8 waves/block
    qkv_gemm_kernel<<<(128 * 48) / 8, 256, 0, stream>>>(Xb, WqkvT, b_qkv,
                                                        Qb, Kb, Vt);
    // 16 heads x 128 q-strips = 2048 waves
    attn_kernel<<<(NHEAD * (S_LEN / 32)) / 8, 256, 0, stream>>>(Qb, Kb, Vt, Ctx);
    // 128 M-strips x 16 N-strips = 2048 waves
    out_gemm_kernel<<<(128 * 16) / 8, 256, 0, stream>>>(Ctx, WoutT, b_out, out);
}
